// DisenGCN_81295140979042
// MI455X (gfx1250) — matrix operand order repaired and checked by reading, not compiled, not run on hardware
//
#include <hip/hip_runtime.h>
#include <hip/hip_bf16.h>

typedef __attribute__((ext_vector_type(16))) __bf16 v16bf;
typedef __attribute__((ext_vector_type(8)))  float  v8f;

#define N_NODES 50000
#define M_EDGES 800000
#define INDIM   512
#define NDIM    128
#define KCH     8
#define DD      16
#define ROUTIT  4
#define NLAYER  3
#define NCLASS  16

// ---------------------------------------------------------------------------
// Pack a row-major [K x ncols] f32 weight matrix into bf16 WMMA B-fragments.
// Fragment layout (per ISA 16-bit B 32x16): lane L holds column n = nt*16+(L&15),
// element e holds k = kt*32 + (e<8 ? (L>>4)*8+e : 16+(L>>4)*8+(e-8)). Stored per-lane contiguous so the
// GEMM kernel loads one 32-byte v16bf per lane.
// frag index = ((kt*ntn + nt)*32 + lane)*16
// ---------------------------------------------------------------------------
__global__ void pack_b_kernel(const float* __restrict__ w, __bf16* __restrict__ frag,
                              int K, int ncols, int ntn) {
    int idx = blockIdx.x * blockDim.x + threadIdx.x;
    int total = (K / 32) * ntn * 32;
    if (idx >= total) return;
    int lane = idx & 31;
    int nt = (idx >> 5) % ntn;
    int kt = (idx >> 5) / ntn;
    int n = nt * 16 + (lane & 15);
    int hb = lane >> 4;
    __bf16* dst = frag + (size_t)idx * 16;
#pragma unroll
    for (int e = 0; e < 16; ++e)
        dst[e] = (__bf16)w[(size_t)(kt * 32 + ((e < 8) ? hb * 8 + e : 16 + hb * 8 + (e - 8))) * ncols + n];
}

// ---------------------------------------------------------------------------
// PCA GEMM: X = relu(feat[50000x512] @ W[512x128] + b).
// One wave per 16-row strip computing ALL 8 N-tiles (full 16x128 output strip):
// the A fragment (f32->bf16 converted in-register) is reused across 8
// back-to-back v_wmma_f32_16x16x32_bf16 per k-step, so feat is read exactly
// once (102 MB) instead of 8x, and cvt work is amortized 8x.
// ---------------------------------------------------------------------------
__global__ void gemm_pca_kernel(const float* __restrict__ A, const __bf16* __restrict__ Bfrag,
                                const float* __restrict__ bias, float* __restrict__ Xout) {
    int wave = (blockIdx.x * blockDim.x + threadIdx.x) >> 5;
    int lane = threadIdx.x & 31;
    const int NTM = N_NODES / 16;   // 3125
    const int NTN = NDIM / 16;      // 8
    if (wave >= NTM) return;
    int mt = wave;
    int mrow  = mt * 16 + (lane & 15);   // A row for this lane
    int khalf = lane >> 4;

    v8f acc[NTN];
#pragma unroll
    for (int nt = 0; nt < NTN; ++nt) {
        float b = bias[nt * 16 + (lane & 15)];
#pragma unroll
        for (int j = 0; j < 8; ++j) acc[nt][j] = b;
    }

    for (int kt = 0; kt < INDIM / 32; ++kt) {
        const float* arow = A + (size_t)mrow * INDIM + kt * 32;
        float4 a0 = *(const float4*)(arow + khalf * 8);
        float4 a1 = *(const float4*)(arow + khalf * 8 + 4);
        float4 a2 = *(const float4*)(arow + 16 + khalf * 8);
        float4 a3 = *(const float4*)(arow + 16 + khalf * 8 + 4);
        v16bf af;
        af[0]=(__bf16)a0.x; af[1]=(__bf16)a0.y; af[2]=(__bf16)a0.z; af[3]=(__bf16)a0.w;
        af[4]=(__bf16)a1.x; af[5]=(__bf16)a1.y; af[6]=(__bf16)a1.z; af[7]=(__bf16)a1.w;
        af[8]=(__bf16)a2.x; af[9]=(__bf16)a2.y; af[10]=(__bf16)a2.z; af[11]=(__bf16)a2.w;
        af[12]=(__bf16)a3.x; af[13]=(__bf16)a3.y; af[14]=(__bf16)a3.z; af[15]=(__bf16)a3.w;
        const __bf16* bbase = Bfrag + ((size_t)kt * NTN * 32 + lane) * 16;
#pragma unroll
        for (int nt = 0; nt < NTN; ++nt) {
            v16bf bf = *(const v16bf*)(bbase + (size_t)nt * 32 * 16);
            acc[nt] = __builtin_amdgcn_wmma_f32_16x16x32_bf16(false, af, false, bf,
                                                              (short)0, acc[nt], false, false);
        }
    }
    // D layout: lane half h = lane>>4, col = nt*16+(lane&15), row = mt*16 + 8*h + j
#pragma unroll
    for (int nt = 0; nt < NTN; ++nt) {
        int col = nt * 16 + (lane & 15);
#pragma unroll
        for (int j = 0; j < 8; ++j) {
            int r = mt * 16 + 8 * khalf + j;
            Xout[(size_t)r * NDIM + col] = fmaxf(acc[nt][j], 0.0f);
        }
    }
}

// ---------------------------------------------------------------------------
// MLP GEMM: logits = relu(X)[50000x128] @ Wm[128x16] + bm  (bf16 WMMA)
// ---------------------------------------------------------------------------
__global__ void gemm_mlp_kernel(const float* __restrict__ Xin, const __bf16* __restrict__ Bfrag,
                                const float* __restrict__ bias, float* __restrict__ logits) {
    int wave = (blockIdx.x * blockDim.x + threadIdx.x) >> 5;
    int lane = threadIdx.x & 31;
    if (wave >= N_NODES / 16) return;
    int mt = wave;
    int mrow  = mt * 16 + (lane & 15);
    int khalf = lane >> 4;

    float b = bias[lane & 15];
    v8f acc;
#pragma unroll
    for (int j = 0; j < 8; ++j) acc[j] = b;

#pragma unroll
    for (int kt = 0; kt < NDIM / 32; ++kt) {
        const float* arow = Xin + (size_t)mrow * NDIM + kt * 32;
        float4 a0 = *(const float4*)(arow + khalf * 8);
        float4 a1 = *(const float4*)(arow + khalf * 8 + 4);
        float4 a2 = *(const float4*)(arow + 16 + khalf * 8);
        float4 a3 = *(const float4*)(arow + 16 + khalf * 8 + 4);
        v16bf af;   // relu fused on load
        af[0]=(__bf16)fmaxf(a0.x,0.f); af[1]=(__bf16)fmaxf(a0.y,0.f);
        af[2]=(__bf16)fmaxf(a0.z,0.f); af[3]=(__bf16)fmaxf(a0.w,0.f);
        af[4]=(__bf16)fmaxf(a1.x,0.f); af[5]=(__bf16)fmaxf(a1.y,0.f);
        af[6]=(__bf16)fmaxf(a1.z,0.f); af[7]=(__bf16)fmaxf(a1.w,0.f);
        af[8]=(__bf16)fmaxf(a2.x,0.f); af[9]=(__bf16)fmaxf(a2.y,0.f);
        af[10]=(__bf16)fmaxf(a2.z,0.f); af[11]=(__bf16)fmaxf(a2.w,0.f);
        af[12]=(__bf16)fmaxf(a3.x,0.f); af[13]=(__bf16)fmaxf(a3.y,0.f);
        af[14]=(__bf16)fmaxf(a3.z,0.f); af[15]=(__bf16)fmaxf(a3.w,0.f);
        v16bf bf = *(const v16bf*)(Bfrag + ((size_t)kt * 32 + lane) * 16);
        acc = __builtin_amdgcn_wmma_f32_16x16x32_bf16(false, af, false, bf,
                                                      (short)0, acc, false, false);
    }
    int col = lane & 15;
#pragma unroll
    for (int j = 0; j < 8; ++j) {
        int r = mt * 16 + 8 * khalf + j;
        logits[(size_t)r * NCLASS + col] = acc[j];
    }
}

// ---------------------------------------------------------------------------
// Per-(node,channel) L2 normalization. Optionally relu on load. Writes out1,
// and optionally out2 (accumulator init = normalized value).
// One thread per (node, channel): 16 floats = 4x float4.
// ---------------------------------------------------------------------------
__global__ void norm_kernel(const float* __restrict__ in, float* __restrict__ out1,
                            float* __restrict__ out2, int do_relu) {
    int idx = blockIdx.x * blockDim.x + threadIdx.x;
    if (idx >= N_NODES * KCH) return;
    const float4* p = (const float4*)(in + (size_t)idx * DD);
    float4 v0 = p[0], v1 = p[1], v2 = p[2], v3 = p[3];
    if (do_relu) {
        v0.x=fmaxf(v0.x,0.f); v0.y=fmaxf(v0.y,0.f); v0.z=fmaxf(v0.z,0.f); v0.w=fmaxf(v0.w,0.f);
        v1.x=fmaxf(v1.x,0.f); v1.y=fmaxf(v1.y,0.f); v1.z=fmaxf(v1.z,0.f); v1.w=fmaxf(v1.w,0.f);
        v2.x=fmaxf(v2.x,0.f); v2.y=fmaxf(v2.y,0.f); v2.z=fmaxf(v2.z,0.f); v2.w=fmaxf(v2.w,0.f);
        v3.x=fmaxf(v3.x,0.f); v3.y=fmaxf(v3.y,0.f); v3.z=fmaxf(v3.z,0.f); v3.w=fmaxf(v3.w,0.f);
    }
    float s = v0.x*v0.x + v0.y*v0.y + v0.z*v0.z + v0.w*v0.w
            + v1.x*v1.x + v1.y*v1.y + v1.z*v1.z + v1.w*v1.w
            + v2.x*v2.x + v2.y*v2.y + v2.z*v2.z + v2.w*v2.w
            + v3.x*v3.x + v3.y*v3.y + v3.z*v3.z + v3.w*v3.w;
    float scale = 1.0f / fmaxf(sqrtf(s), 1e-12f);
    v0.x*=scale; v0.y*=scale; v0.z*=scale; v0.w*=scale;
    v1.x*=scale; v1.y*=scale; v1.z*=scale; v1.w*=scale;
    v2.x*=scale; v2.y*=scale; v2.z*=scale; v2.w*=scale;
    v3.x*=scale; v3.y*=scale; v3.z*=scale; v3.w*=scale;
    float4* q1 = (float4*)(out1 + (size_t)idx * DD);
    q1[0]=v0; q1[1]=v1; q1[2]=v2; q1[3]=v3;
    if (out2) {
        float4* q2 = (float4*)(out2 + (size_t)idx * DD);
        q2[0]=v0; q2[1]=v1; q2[2]=v2; q2[3]=v3;
    }
}

// ---------------------------------------------------------------------------
// Routing edge kernel: one wave (32 lanes) per edge, lane l owns dims [4l,4l+4).
// Channel k occupies lanes 4k..4k+3 (dd=16). Channel dot via xor-shuffles
// (strides 1,2), softmax over 8 channels via xor-shuffles (strides 4,8,16),
// then atomic scatter-add of p_k * z into the target-node accumulator.
// All traffic is L2-resident (node arrays = 25.6 MB << 192 MB L2).
// ---------------------------------------------------------------------------
__global__ void edge_kernel(const int* __restrict__ src, const int* __restrict__ trg,
                            const float* __restrict__ Z, const float* __restrict__ C,
                            float* __restrict__ Acc) {
    int gid = blockIdx.x * blockDim.x + threadIdx.x;
    int e = gid >> 5;
    int lane = threadIdx.x & 31;
    if (e >= M_EDGES) return;
    int s = src[e];
    int t = trg[e];
    float4 z = ((const float4*)(Z + (size_t)s * NDIM))[lane];
    float4 c = ((const float4*)(C + (size_t)t * NDIM))[lane];
    float d = z.x*c.x + z.y*c.y + z.z*c.z + z.w*c.w;
    d += __shfl_xor(d, 1, 32);
    d += __shfl_xor(d, 2, 32);                      // channel dot, replicated in 4-lane group
    float mx = d;
    mx = fmaxf(mx, __shfl_xor(mx, 4, 32));
    mx = fmaxf(mx, __shfl_xor(mx, 8, 32));
    mx = fmaxf(mx, __shfl_xor(mx, 16, 32));         // max over 8 channels
    float ex = __expf(d - mx);
    float se = ex;
    se += __shfl_xor(se, 4, 32);
    se += __shfl_xor(se, 8, 32);
    se += __shfl_xor(se, 16, 32);                   // sum of exp over 8 channels
    float p = ex / se;
    float* dst = Acc + (size_t)t * NDIM + lane * 4;
    atomicAdd(dst + 0, p * z.x);
    atomicAdd(dst + 1, p * z.y);
    atomicAdd(dst + 2, p * z.z);
    atomicAdd(dst + 3, p * z.w);
}

// ---------------------------------------------------------------------------
// Row-wise log_softmax over 16 classes, one thread per node.
// ---------------------------------------------------------------------------
__global__ void logsoftmax_kernel(const float* __restrict__ logits, float* __restrict__ out) {
    int n = blockIdx.x * blockDim.x + threadIdx.x;
    if (n >= N_NODES) return;
    const float4* p = (const float4*)(logits + (size_t)n * NCLASS);
    float4 v0 = p[0], v1 = p[1], v2 = p[2], v3 = p[3];
    float v[16] = { v0.x,v0.y,v0.z,v0.w, v1.x,v1.y,v1.z,v1.w,
                    v2.x,v2.y,v2.z,v2.w, v3.x,v3.y,v3.z,v3.w };
    float mx = v[0];
#pragma unroll
    for (int i = 1; i < 16; ++i) mx = fmaxf(mx, v[i]);
    float se = 0.0f;
#pragma unroll
    for (int i = 0; i < 16; ++i) se += __expf(v[i] - mx);
    float lse = mx + logf(se);
    float4* q = (float4*)(out + (size_t)n * NCLASS);
    float4 o0 = { v[0]-lse, v[1]-lse, v[2]-lse, v[3]-lse };
    float4 o1 = { v[4]-lse, v[5]-lse, v[6]-lse, v[7]-lse };
    float4 o2 = { v[8]-lse, v[9]-lse, v[10]-lse, v[11]-lse };
    float4 o3 = { v[12]-lse, v[13]-lse, v[14]-lse, v[15]-lse };
    q[0]=o0; q[1]=o1; q[2]=o2; q[3]=o3;
}

// ---------------------------------------------------------------------------
extern "C" void kernel_launch(void* const* d_in, const int* in_sizes, int n_in,
                              void* d_out, int out_size, void* d_ws, size_t ws_size,
                              hipStream_t stream) {
    (void)in_sizes; (void)n_in; (void)out_size; (void)ws_size;
    const float* feat   = (const float*)d_in[0];
    const int*   srctrg = (const int*)d_in[1];
    const float* pca_w  = (const float*)d_in[2];
    const float* pca_b  = (const float*)d_in[3];
    const float* mlp_w  = (const float*)d_in[4];
    const float* mlp_b  = (const float*)d_in[5];
    const int* src = srctrg;
    const int* trg = srctrg + M_EDGES;

    // Workspace carve-out (256B aligned)
    char* ws = (char*)d_ws;
    size_t off = 0;
    auto carve = [&](size_t bytes) -> void* {
        void* p = ws + off;
        off = (off + bytes + 255) & ~(size_t)255;
        return p;
    };
    float*  XN     = (float*)carve((size_t)N_NODES * NDIM * sizeof(float));  // fixed z source
    float*  CN     = (float*)carve((size_t)N_NODES * NDIM * sizeof(float));  // normalized c
    float*  CACC   = (float*)carve((size_t)N_NODES * NDIM * sizeof(float));  // accumulator
    float*  logits = (float*)carve((size_t)N_NODES * NCLASS * sizeof(float));
    __bf16* pcaF   = (__bf16*)carve((size_t)(INDIM/32) * (NDIM/16) * 32 * 16 * sizeof(__bf16));
    __bf16* mlpF   = (__bf16*)carve((size_t)(NDIM/32) * 1 * 32 * 16 * sizeof(__bf16));

    // Pre-swizzle B matrices into WMMA fragment layout (tiny)
    {
        int total = (INDIM / 32) * (NDIM / 16) * 32;   // 4096
        pack_b_kernel<<<(total + 255) / 256, 256, 0, stream>>>(pca_w, pcaF, INDIM, NDIM, NDIM / 16);
    }
    {
        int total = (NDIM / 32) * 1 * 32;              // 128
        pack_b_kernel<<<1, 128, 0, stream>>>(mlp_w, mlpF, NDIM, NCLASS, 1);
    }

    // PCA projection + relu -> CACC  (one wave per 16-row strip, all 8 N-tiles)
    {
        int waves = N_NODES / 16;                      // 3125
        gemm_pca_kernel<<<(waves * 32 + 255) / 256, 256, 0, stream>>>(feat, pcaF, pca_b, CACC);
    }

    const int normBlocks = (N_NODES * KCH + 255) / 256;
    const int edgeBlocks = (M_EDGES * 32 + 255) / 256;

    for (int layer = 0; layer < NLAYER; ++layer) {
        // layer-start: (relu for layers>0; layer0 input already relu'd) + normalize
        // -> XN (fixed neighbor-message source) and CACC (accumulator init)
        norm_kernel<<<normBlocks, 256, 0, stream>>>(CACC, XN, CACC, layer > 0 ? 1 : 0);
        for (int t = 0; t < ROUTIT; ++t) {
            const float* csrc = (t == 0) ? XN : CN;
            edge_kernel<<<edgeBlocks, 256, 0, stream>>>(src, trg, XN, csrc, CACC);
            if (t < ROUTIT - 1) {
                // c = l2norm(c); write CN (gather source) and CACC (next accumulator base)
                norm_kernel<<<normBlocks, 256, 0, stream>>>(CACC, CN, CACC, 0);
            }
        }
        // un-normalized result stays in CACC -> next layer (relu fused into norm) or MLP
    }

    // Final MLP (relu fused on A load) + log_softmax
    {
        int waves = N_NODES / 16;                      // 3125
        gemm_mlp_kernel<<<(waves * 32 + 255) / 256, 256, 0, stream>>>(CACC, mlpF, mlp_b, logits);
    }
    logsoftmax_kernel<<<(N_NODES + 255) / 256, 256, 0, stream>>>(logits, (float*)d_out);
}
